// StochasticLSTMPredictor_18863496364523
// MI455X (gfx1250) — matrix operand order repaired and checked by reading, not compiled, not run on hardware
//
#include <hip/hip_runtime.h>
#include <hip/hip_bf16.h>

typedef __attribute__((ext_vector_type(16))) _Float16 v16h;
typedef __attribute__((ext_vector_type(8)))  _Float16 v8h;
typedef __attribute__((ext_vector_type(8)))  float    v8f;

#define B_   256
#define T_   512
#define I_   128
#define H_   256
#define L_   64
#define HPR_ 50
#define HP_  64      // HPR padded to WMMA N granule
#define XZW_ 448     // [ x(128) | h(256) | z(64) ] concat row width
#define BT_  16      // batch rows per workgroup (WMMA M)

// f16 weight workspace layout (element offsets, all 4096-multiples -> 32B aligned)
#define OFF_WQM1 0        // 64 x 384
#define OFF_WQL1 24576    // 64 x 384
#define OFF_WPM1 49152    // 64 x 256
#define OFF_WPL1 65536    // 64 x 256
#define OFF_WQM2 81920    // 64 x 64
#define OFF_WQL2 86016    // 64 x 64
#define OFF_WPM2 90112    // 64 x 64
#define OFF_WPL2 94208    // 64 x 64
#define OFF_WH1  98304    // 64 x 448
#define OFF_WH2  126976   // 256 x 64
#define OFF_WO   143360   // 128 x 320

// d_out layout: x_hat | p_mu | p_lv | q_mu | q_lv
#define OUT_PMU  ((size_t)B_ * T_ * I_)
#define OUT_PLV  (OUT_PMU + (size_t)B_ * T_ * L_)
#define OUT_QMU  (OUT_PLV + (size_t)B_ * T_ * L_)
#define OUT_QLV  (OUT_QMU + (size_t)B_ * T_ * L_)

// ---------------- fragment loaders (contiguous k order k = 16*(lane/16) + s, wave32) ----------------

// A 16x32 f16: lane<16 -> row=lane, K k..k+15; lane>=16 -> row=lane-16, K k+16..k+31
__device__ __forceinline__ v16h load_a_lds(const _Float16* base, int ld, int kbase, int lane) {
    const int row = lane & 15;
    const int ko  = (lane >> 4) * 16;
    const _Float16* p = base + row * ld + kbase + ko;
    union { v16h v; v8h h[2]; } a;
    a.h[0] = *(const v8h*)(p);
    a.h[1] = *(const v8h*)(p + 8);
    return a.v;
}

// B 32x16 f16 (stored as W[n][k], B = W^T): lane<16 -> col=lane, K k..k+15; lane>=16 -> col=lane-16, K k+16..k+31
__device__ __forceinline__ v16h load_b_ws(const _Float16* W, int ldk, int nbase, int kbase, int lane) {
    const int n  = lane & 15;
    const int ko = (lane >> 4) * 16;
    return *(const v16h*)(W + (size_t)(nbase + n) * ldk + kbase + ko);
}

__device__ __forceinline__ v8f wmma32(v8f c, v16h a, v16h b) {
    return __builtin_amdgcn_wmma_f32_16x16x32_f16(false, a, false, b, (short)0, c, false, false);
}

#define VZERO {0.f,0.f,0.f,0.f,0.f,0.f,0.f,0.f}

// ---------------- weight f32 -> f16 convert + pad pre-pass ----------------
__global__ void convert_pad_kernel(const float* __restrict__ src, _Float16* __restrict__ dst,
                                   int N, int K, int Kpad, int total) {
    int i = blockIdx.x * blockDim.x + threadIdx.x;
    if (i >= total) return;
    int n = i / Kpad, k = i - n * Kpad;
    float v = (k < K && n < N) ? src[n * K + k] : 0.f;
    dst[i] = (_Float16)v;
}

// ---------------- persistent scan kernel: one WG per 16 batch rows ----------------
__global__ void __launch_bounds__(256, 1)
vrnn_scan_kernel(const float* __restrict__ x, const float* __restrict__ hidden,
                 const float* __restrict__ noise, const _Float16* __restrict__ ws,
                 const float* __restrict__ bqm1, const float* __restrict__ bqm2,
                 const float* __restrict__ bql1, const float* __restrict__ bql2,
                 const float* __restrict__ bpm1, const float* __restrict__ bpm2,
                 const float* __restrict__ bpl1, const float* __restrict__ bpl2,
                 const float* __restrict__ bh1,  const float* __restrict__ bh2,
                 const float* __restrict__ bo,   float* __restrict__ out)
{
    __shared__ __align__(32) _Float16 xz[BT_ * XZW_];     // [x|h|z] f16, 14.0 KB
    __shared__ __align__(32) _Float16 a1[4][BT_ * HP_];   // L1 activations (q_mu,q_lv,p_mu,p_lv)
    __shared__ __align__(32) float    uq[2][BT_ * L_];    // q_mu, q_lv (f32, for z)
    __shared__ __align__(32) _Float16 a1h[BT_ * HP_];     // h-net L1 activation
    __shared__ __align__(32) _Float16 hbuf[BT_ * H_];     // h_next staging

    const int tid  = threadIdx.x;
    const int lane = tid & 31;
    const int wave = tid >> 5;          // 8 waves
    const int b0   = blockIdx.x * BT_;

    const _Float16* L1W[4] = { ws+OFF_WQM1, ws+OFF_WQL1, ws+OFF_WPM1, ws+OFF_WPL1 };
    const _Float16* L2W[4] = { ws+OFF_WQM2, ws+OFF_WQL2, ws+OFF_WPM2, ws+OFF_WPL2 };
    const float*    B1[4]  = { bqm1, bql1, bpm1, bpl1 };
    const float*    B2[4]  = { bqm2, bql2, bpm2, bpl2 };
    const size_t    OB[4]  = { OUT_QMU, OUT_QLV, OUT_PMU, OUT_PLV };
    const _Float16* Wh1p = ws + OFF_WH1;
    const _Float16* Wh2p = ws + OFF_WH2;
    const _Float16* Wop  = ws + OFF_WO;

    for (int t = 0; t < T_; ++t) {
        // ---- phase A: refresh xz = [x_t | h] (z slot filled later) ----
        for (int i = tid; i < BT_ * H_; i += 256) {
            int row = i >> 8, col = i & (H_ - 1);
            _Float16 hv = (t == 0) ? (_Float16)hidden[(size_t)(b0 + row) * H_ + col] : hbuf[i];
            xz[row * XZW_ + I_ + col] = hv;
        }
        for (int i = tid; i < BT_ * I_; i += 256) {
            int row = i >> 7, col = i & (I_ - 1);
            xz[row * XZW_ + col] = (_Float16)x[((size_t)(b0 + row) * T_ + t) * I_ + col];
        }
        if (t + 1 < T_ && tid < BT_) {     // pull next timestep's x tile toward L2/L0
            const float* p = &x[((size_t)(b0 + tid) * T_ + t + 1) * I_];
            __builtin_prefetch(p, 0, 1);
            __builtin_prefetch(p + 64, 0, 1);
        }
        __syncthreads();

        // ---- phase B: four L1 MLPs (relu(x@W1.T+b1)); 2 waves per MLP, 2 N-tiles each ----
        {
            const int mlp = wave >> 1;
            const int K   = (mlp < 2) ? (I_ + H_) : H_;     // 384 or 256
            const int co  = (mlp < 2) ? 0 : I_;             // q reads [x|h], p reads [h]
            const _Float16* W  = L1W[mlp];
            const float*    b1 = B1[mlp];
            for (int nt = 2 * (wave & 1); nt < 2 * (wave & 1) + 2; ++nt) {
                v8f acc = VZERO;
                for (int k0 = 0; k0 < K; k0 += 32)
                    acc = wmma32(acc, load_a_lds(xz, XZW_, co + k0, lane),
                                      load_b_ws(W, K, nt * 16, k0, lane));
                const int col = nt * 16 + (lane & 15);
                const float bv = (col < HPR_) ? b1[col] : 0.f;
                const int r0 = (lane >> 4) * 8;
                #pragma unroll
                for (int g = 0; g < 8; ++g) {
                    float v = acc[g] + bv;
                    a1[mlp][(r0 + g) * HP_ + col] = (_Float16)(v > 0.f ? v : 0.f);
                }
            }
        }
        __syncthreads();

        // ---- phase C: four L2 heads -> q_mu/q_lv/p_mu/p_lv [16x64] f32, write out ----
        {
            const int mlp = wave >> 1;
            const _Float16* W2 = L2W[mlp];
            const float*    b2 = B2[mlp];
            for (int nt = 2 * (wave & 1); nt < 2 * (wave & 1) + 2; ++nt) {
                v8f acc = VZERO;
                #pragma unroll
                for (int k0 = 0; k0 < HP_; k0 += 32)
                    acc = wmma32(acc, load_a_lds(a1[mlp], HP_, k0, lane),
                                      load_b_ws(W2, HP_, nt * 16, k0, lane));
                const int col = nt * 16 + (lane & 15);
                const float bv = b2[col];
                const int r0 = (lane >> 4) * 8;
                #pragma unroll
                for (int g = 0; g < 8; ++g) {
                    const int row = r0 + g;
                    float v = acc[g] + bv;
                    out[OB[mlp] + ((size_t)(b0 + row) * T_ + t) * L_ + col] = v;
                    if (mlp == 0)      uq[0][row * L_ + col] = v;
                    else if (mlp == 1) uq[1][row * L_ + col] = v;
                }
            }
        }
        __syncthreads();

        // ---- z = q_mu + exp(0.5*q_lv)*eps, into xz z-slot ----
        for (int i = tid; i < BT_ * L_; i += 256) {
            int row = i >> 6, col = i & (L_ - 1);
            float zv = uq[0][i] + __expf(0.5f * uq[1][i]) *
                       noise[((size_t)(b0 + row) * T_ + t) * L_ + col];
            xz[row * XZW_ + (I_ + H_) + col] = (_Float16)zv;
        }
        __syncthreads();

        // ---- phase D1: h-net L1 over [x|h|z], K=448; waves 0..3 take one N-tile each ----
        if (wave < 4) {
            const int nt = wave;
            v8f acc = VZERO;
            for (int k0 = 0; k0 < XZW_; k0 += 32)
                acc = wmma32(acc, load_a_lds(xz, XZW_, k0, lane),
                                  load_b_ws(Wh1p, XZW_, nt * 16, k0, lane));
            const int col = nt * 16 + (lane & 15);
            const float bv = (col < HPR_) ? bh1[col] : 0.f;
            const int r0 = (lane >> 4) * 8;
            #pragma unroll
            for (int g = 0; g < 8; ++g) {
                float v = acc[g] + bv;
                a1h[(r0 + g) * HP_ + col] = (_Float16)(v > 0.f ? v : 0.f);
            }
        }
        __syncthreads();

        // ---- phase D2: h_next = a1h @ Wh2.T + bh2 (N=256, 16 tiles / 8 waves) ----
        for (int nt = wave * 2; nt < wave * 2 + 2; ++nt) {
            v8f acc = VZERO;
            #pragma unroll
            for (int k0 = 0; k0 < HP_; k0 += 32)
                acc = wmma32(acc, load_a_lds(a1h, HP_, k0, lane),
                                  load_b_ws(Wh2p, HP_, nt * 16, k0, lane));
            const int col = nt * 16 + (lane & 15);
            const float bv = bh2[col];
            const int r0 = (lane >> 4) * 8;
            #pragma unroll
            for (int g = 0; g < 8; ++g)
                hbuf[(r0 + g) * H_ + col] = (_Float16)(acc[g] + bv);
        }

        // ---- phase E: out = sigmoid([h|z] @ Wo.T + bo); reads OLD h still in xz ----
        {
            const int nt = wave;                 // N=128 -> 8 tiles, 1 per wave
            v8f acc = VZERO;
            for (int k0 = 0; k0 < (H_ + L_); k0 += 32)
                acc = wmma32(acc, load_a_lds(xz, XZW_, I_ + k0, lane),
                                  load_b_ws(Wop, H_ + L_, nt * 16, k0, lane));
            const int col = nt * 16 + (lane & 15);
            const float bv = bo[col];
            const int r0 = (lane >> 4) * 8;
            #pragma unroll
            for (int g = 0; g < 8; ++g) {
                float v = acc[g] + bv;
                v = 1.f / (1.f + __expf(-v));
                out[((size_t)(b0 + r0 + g) * T_ + t) * I_ + col] = v;
            }
        }
        __syncthreads();   // protect xz (old h) and hbuf before next iteration
    }
}

extern "C" void kernel_launch(void* const* d_in, const int* in_sizes, int n_in,
                              void* d_out, int out_size, void* d_ws, size_t ws_size,
                              hipStream_t stream) {
    (void)in_sizes; (void)n_in; (void)out_size; (void)ws_size;
    const float* x      = (const float*)d_in[0];
    const float* hidden = (const float*)d_in[1];
    const float* noise  = (const float*)d_in[2];
    const float* Wpm1 = (const float*)d_in[3];  const float* bpm1 = (const float*)d_in[4];
    const float* Wpm2 = (const float*)d_in[5];  const float* bpm2 = (const float*)d_in[6];
    const float* Wpl1 = (const float*)d_in[7];  const float* bpl1 = (const float*)d_in[8];
    const float* Wpl2 = (const float*)d_in[9];  const float* bpl2 = (const float*)d_in[10];
    const float* Wqm1 = (const float*)d_in[11]; const float* bqm1 = (const float*)d_in[12];
    const float* Wqm2 = (const float*)d_in[13]; const float* bqm2 = (const float*)d_in[14];
    const float* Wql1 = (const float*)d_in[15]; const float* bql1 = (const float*)d_in[16];
    const float* Wql2 = (const float*)d_in[17]; const float* bql2 = (const float*)d_in[18];
    const float* Wh1  = (const float*)d_in[19]; const float* bh1  = (const float*)d_in[20];
    const float* Wh2  = (const float*)d_in[21]; const float* bh2  = (const float*)d_in[22];
    const float* Wo   = (const float*)d_in[23]; const float* bo   = (const float*)d_in[24];
    _Float16* ws = (_Float16*)d_ws;
    float* out = (float*)d_out;

    auto conv = [&](const float* src, size_t off, int N, int K, int Npad, int Kpad) {
        int total = Npad * Kpad;
        convert_pad_kernel<<<(total + 255) / 256, 256, 0, stream>>>(src, ws + off, N, K, Kpad, total);
    };
    conv(Wqm1, OFF_WQM1,  50, 384,  64, 384);
    conv(Wql1, OFF_WQL1,  50, 384,  64, 384);
    conv(Wpm1, OFF_WPM1,  50, 256,  64, 256);
    conv(Wpl1, OFF_WPL1,  50, 256,  64, 256);
    conv(Wqm2, OFF_WQM2,  64,  50,  64,  64);
    conv(Wql2, OFF_WQL2,  64,  50,  64,  64);
    conv(Wpm2, OFF_WPM2,  64,  50,  64,  64);
    conv(Wpl2, OFF_WPL2,  64,  50,  64,  64);
    conv(Wh1,  OFF_WH1,   50, 448,  64, 448);
    conv(Wh2,  OFF_WH2,  256,  50, 256,  64);
    conv(Wo,   OFF_WO,   128, 320, 128, 320);

    vrnn_scan_kernel<<<B_ / BT_, 256, 0, stream>>>(
        x, hidden, noise, ws,
        bqm1, bqm2, bql1, bql2, bpm1, bpm2, bpl1, bpl2,
        bh1, bh2, bo, out);
}
